// MLP_22565758173772
// MI455X (gfx1250) — compile-verified, matrix operand order checked by reading, not run on hardware
//
#include <hip/hip_runtime.h>

// ---------------------------------------------------------------------------
// Types
// ---------------------------------------------------------------------------
typedef _Float16 f16;
typedef __attribute__((ext_vector_type(16))) _Float16 v16h;
typedef __attribute__((ext_vector_type(8)))  _Float16 v8h;
typedef __attribute__((ext_vector_type(8)))  float    v8f;
typedef int v4i_vs __attribute__((vector_size(16)));   // matches builtin param
typedef __attribute__((address_space(1))) v4i_vs as1_v4i;
typedef __attribute__((address_space(3))) v4i_vs as3_v4i;

// ---------------------------------------------------------------------------
// CDNA5 async global->LDS copy path (ASYNCcnt-tracked DMA), with a
// synchronous VGPR-staging fallback if the toolchain lacks the builtins.
// ---------------------------------------------------------------------------
#if __has_builtin(__builtin_amdgcn_global_load_async_to_lds_b128)
#define USE_ASYNC_COPY 1
#else
#define USE_ASYNC_COPY 0
#endif

#if USE_ASYNC_COPY
#if __has_builtin(__builtin_amdgcn_s_wait_asynccnt)
#define ASYNC_WAIT(n) __builtin_amdgcn_s_wait_asynccnt(n)
#else
#define ASYNC_WAIT(n) asm volatile("s_wait_asynccnt %0" ::"n"(n) : "memory")
#endif
#else
#define ASYNC_WAIT(n) ((void)0)
#endif

__device__ __forceinline__ void copy16B_to_lds(const f16* g, f16* l) {
#if USE_ASYNC_COPY
    __builtin_amdgcn_global_load_async_to_lds_b128(
        (as1_v4i*)g, (as3_v4i*)l, /*offset=*/0, /*cpol=*/0);
#else
    *(v8h*)l = *(const v8h*)g;
#endif
}

// NF4 codebook
__device__ __constant__ float NF4_LUT[16] = {
    -1.0f, -0.6961928009986877f, -0.5250730514526367f, -0.39491748809814453f,
    -0.28444138169288635f, -0.18477343022823334f, -0.09105003625154495f, 0.0f,
    0.07958029955625534f, 0.16093020141124725f, 0.24611230194568634f,
    0.33791524171829224f, 0.44070982933044434f, 0.5626170039176941f,
    0.6961928009986877f, 1.0f};

// ---------------------------------------------------------------------------
// NF4 dequant: int32 codes [R,C] + fp32 absmax [R, C/64] -> fp16 [R,C]
// ---------------------------------------------------------------------------
__global__ __launch_bounds__(256)
void dequant_nf4_kernel(const int* __restrict__ codes,
                        const float* __restrict__ absmax,
                        f16* __restrict__ out,
                        int C, long long total) {
    long long base = ((long long)blockIdx.x * blockDim.x + threadIdx.x) * 8ll;
    if (base >= total) return;
    long long row = base / C;
    int col = (int)(base - row * (long long)C);
    float s = absmax[row * (long long)(C >> 6) + (col >> 6)];

    int4 c0 = *(const int4*)(codes + base);
    int4 c1 = *(const int4*)(codes + base + 4);
    v8h o;
    o[0] = (f16)(NF4_LUT[c0.x & 15] * s);
    o[1] = (f16)(NF4_LUT[c0.y & 15] * s);
    o[2] = (f16)(NF4_LUT[c0.z & 15] * s);
    o[3] = (f16)(NF4_LUT[c0.w & 15] * s);
    o[4] = (f16)(NF4_LUT[c1.x & 15] * s);
    o[5] = (f16)(NF4_LUT[c1.y & 15] * s);
    o[6] = (f16)(NF4_LUT[c1.z & 15] * s);
    o[7] = (f16)(NF4_LUT[c1.w & 15] * s);
    *(v8h*)(out + base) = o;
}

// ---------------------------------------------------------------------------
// WMMA fragment load from an LDS tile (CDNA5 16-bit A/B layout):
//   lanes 0-15 : row = lane,    K chunks {0..7} and {16..23}
//   lanes 16-31: row = lane-16, K chunks {8..15} and {24..31}
// ---------------------------------------------------------------------------
__device__ __forceinline__ v16h load_frag(const f16* p, int stride) {
    int lane = threadIdx.x & 31;
    int row = lane & 15;
    int kbase = (lane & 16) ? 8 : 0;
    const f16* q = p + row * stride + kbase;
    v8h c0 = *(const v8h*)(q);
    v8h c1 = *(const v8h*)(q + 16);
    return __builtin_shufflevector(c0, c1, 0, 1, 2, 3, 4, 5, 6, 7,
                                           8, 9, 10, 11, 12, 13, 14, 15);
}

__device__ __forceinline__ v8f wmma_f16(v16h a, v16h b, v8f c) {
    return __builtin_amdgcn_wmma_f32_16x16x32_f16(
        false, a, false, b, (short)0, c, false, false);
}

// ---------------------------------------------------------------------------
// Tiled WMMA GEMM:  C[M,N] = A[M,K] * B[N,K]^T   (fp16 in, fp32 accum)
// GATED: also Bu[N,K];  C = silu(A*Bg^T) * (A*Bu^T)
//
// Block tile 128(M) x 128(N), K-step 64, double-buffered LDS filled by
// async global->LDS DMA. 256 threads = 8 waves (2M x 4N); each wave owns a
// 64x32 sub-tile = 4x2 WMMA tiles (x2 accumulator sets when GATED), giving
// 16 WMMAs per 8 fragment loads per 32-K sub-step.
// ---------------------------------------------------------------------------
#define KSTEP 64
#define LDST  72  // 64 halves + 8 pad: 144B rows (16B-aligned, bank-staggered)

template <bool GATED>
__global__ __launch_bounds__(256)
void wmma_gemm_kernel(const f16* __restrict__ A,   // [M,K]
                      const f16* __restrict__ Bg,  // [N,K]
                      const f16* __restrict__ Bu,  // [N,K] (GATED only)
                      f16* __restrict__ C,         // [M,N]
                      int M, int N, int K) {
    __shared__ f16 ldsA[2][128 * LDST];               // 2 x 18 KB
    __shared__ f16 ldsBg[2][128 * LDST];              // 2 x 18 KB
    __shared__ f16 ldsBu[GATED ? 2 : 1][GATED ? 128 * LDST : 8];

    const int t = threadIdx.x;
    const int wave = t >> 5;
    const int waveM = wave >> 2;  // 0..1 -> 64-row slice
    const int waveN = wave & 3;   // 0..3 -> 32-col slice
    const int blockN0 = blockIdx.x * 128;
    const int blockM0 = blockIdx.y * 128;

    constexpr int NCH = GATED ? 12 : 8;  // async-copy insts per tile per wave

    v8f accG[8], accU[8];
    v8f zero = {};
#pragma unroll
    for (int i = 0; i < 8; ++i) { accG[i] = zero; accU[i] = zero; }

    // Chunk map: 16B (8-half) chunks; each 128x64-half tile = 1024 chunks
    // = 4 chunks per thread per tensor.
    auto issueTile = [&](int kt, int b) {
        const int k0 = kt * KSTEP;
#pragma unroll
        for (int i = 0; i < 4; ++i) {
            int c = t + i * 256;
            int row = c >> 3, col = (c & 7) * 8;
            copy16B_to_lds(A + (size_t)(blockM0 + row) * K + k0 + col,
                           &ldsA[b][row * LDST + col]);
            copy16B_to_lds(Bg + (size_t)(blockN0 + row) * K + k0 + col,
                           &ldsBg[b][row * LDST + col]);
            if constexpr (GATED)
                copy16B_to_lds(Bu + (size_t)(blockN0 + row) * K + k0 + col,
                               &ldsBu[b][row * LDST + col]);
        }
    };

    const int KT = K / KSTEP;
    issueTile(0, 0);

    for (int kt = 0; kt < KT; ++kt) {
        const int cur = kt & 1;
        if (kt + 1 < KT) {
            issueTile(kt + 1, (kt + 1) & 1);
            ASYNC_WAIT(NCH);   // tile kt's copies (older, in-order) have landed
        } else {
            ASYNC_WAIT(0);
        }
        __syncthreads();       // make tile kt visible to all waves

        const f16* bufA = ldsA[cur];
        const f16* bufG = ldsBg[cur];
        const f16* bufU = ldsBu[GATED ? cur : 0];
#pragma unroll
        for (int ks = 0; ks < 2; ++ks) {
            const int ko = ks * 32;
            v16h aF[4];
#pragma unroll
            for (int i = 0; i < 4; ++i)
                aF[i] = load_frag(&bufA[(waveM * 64 + i * 16) * LDST + ko], LDST);
            v16h gF0 = load_frag(&bufG[(waveN * 32 + 0) * LDST + ko], LDST);
            v16h gF1 = load_frag(&bufG[(waveN * 32 + 16) * LDST + ko], LDST);
#pragma unroll
            for (int i = 0; i < 4; ++i) {
                accG[i * 2 + 0] = wmma_f16(aF[i], gF0, accG[i * 2 + 0]);
                accG[i * 2 + 1] = wmma_f16(aF[i], gF1, accG[i * 2 + 1]);
            }
            if constexpr (GATED) {
                v16h uF0 = load_frag(&bufU[(waveN * 32 + 0) * LDST + ko], LDST);
                v16h uF1 = load_frag(&bufU[(waveN * 32 + 16) * LDST + ko], LDST);
#pragma unroll
                for (int i = 0; i < 4; ++i) {
                    accU[i * 2 + 0] = wmma_f16(aF[i], uF0, accU[i * 2 + 0]);
                    accU[i * 2 + 1] = wmma_f16(aF[i], uF1, accU[i * 2 + 1]);
                }
            }
        }
        __syncthreads();       // all reads done before buffer is re-filled
    }

    // Epilogue: C/D layout -> VGPR r : M = r (+8 for lanes 16-31); lane&15 : N
    const int lane = t & 31;
    const int nIdx = lane & 15;
    const int mHi = (lane >> 4) << 3;
#pragma unroll
    for (int mt = 0; mt < 4; ++mt) {
#pragma unroll
        for (int nt = 0; nt < 2; ++nt) {
            v8f g = accG[mt * 2 + nt];
            v8f u = accU[mt * 2 + nt];
            size_t mBase = (size_t)blockM0 + waveM * 64 + mt * 16 + mHi;
            size_t nBase = (size_t)blockN0 + waveN * 32 + nt * 16 + nIdx;
#pragma unroll
            for (int r = 0; r < 8; ++r) {
                float val;
                if constexpr (GATED) {
                    float gv = g[r];
                    val = (gv / (1.0f + __expf(-gv))) * u[r];  // silu(g)*u
                } else {
                    val = g[r];
                }
                C[(mBase + r) * (size_t)N + nBase] = (f16)val;
            }
        }
    }
}

// ---------------------------------------------------------------------------
// Launch:  y = dequant(Wd) @ ( silu(x @ dequant(Wg)^T) * (x @ dequant(Wu)^T) )
// ---------------------------------------------------------------------------
extern "C" void kernel_launch(void* const* d_in, const int* in_sizes, int n_in,
                              void* d_out, int out_size, void* d_ws, size_t ws_size,
                              hipStream_t stream) {
    const f16*   x  = (const f16*)d_in[0];
    const int*   gc = (const int*)d_in[1];
    const float* ga = (const float*)d_in[2];
    const int*   uc = (const int*)d_in[3];
    const float* ua = (const float*)d_in[4];
    const int*   dc = (const int*)d_in[5];
    const float* da = (const float*)d_in[6];

    const int hd = 2048;
    const int m  = in_sizes[1] / hd;        // 8192
    const int tokens = in_sizes[0] / hd;    // 16384

    // Workspace: [Wg fp16 | Wu fp16 | Wd fp16 | h fp16]
    char* ws = (char*)d_ws;
    size_t wbytes = (size_t)m * hd * sizeof(f16);   // 32 MB each
    f16* wg = (f16*)(ws);
    f16* wu = (f16*)(ws + wbytes);
    f16* wd = (f16*)(ws + 2 * wbytes);
    f16* h  = (f16*)(ws + 3 * wbytes);              // [tokens, m] = 256 MB

    long long wtot = (long long)m * hd;
    int dqBlocks = (int)((wtot / 8 + 255) / 256);
    dequant_nf4_kernel<<<dqBlocks, 256, 0, stream>>>(gc, ga, wg, hd, wtot);
    dequant_nf4_kernel<<<dqBlocks, 256, 0, stream>>>(uc, ua, wu, hd, wtot);
    dequant_nf4_kernel<<<dqBlocks, 256, 0, stream>>>(dc, da, wd, m, wtot);

    // Fused gate+up GEMM with SwiGLU epilogue: h = silu(x Wg^T) * (x Wu^T)
    dim3 g1(m / 128, tokens / 128);
    wmma_gemm_kernel<true><<<g1, 256, 0, stream>>>(x, wg, wu, h, tokens, m, hd);

    // Down projection: y = h Wd^T
    dim3 g2(hd / 128, tokens / 128);
    wmma_gemm_kernel<false><<<g2, 256, 0, stream>>>(h, wd, nullptr, (f16*)d_out,
                                                    tokens, hd, m);
}
